// AdvancedOptimizedAttention_29764123361368
// MI455X (gfx1250) — matrix operand order repaired and checked by reading, not compiled, not run on hardware
//
#include <hip/hip_runtime.h>

// ---------------------------------------------------------------------------
// MHA forward for B=4, S=2048, D=1024, H=16, HD=64 on gfx1250 (CDNA5, wave32)
// All matmuls use v_wmma_f32_16x16x32_f16 (f16 inputs, f32 accumulate).
// GEMMs use 32x64 per-wave register blocking: each B fragment feeds 2 WMMAs.
// ---------------------------------------------------------------------------

typedef __attribute__((ext_vector_type(16))) _Float16 v16h;
typedef __attribute__((ext_vector_type(8)))  _Float16 v8h;
typedef __attribute__((ext_vector_type(8)))  float    v8f;

#define WMMA_F32_F16(a, b, c) \
  __builtin_amdgcn_wmma_f32_16x16x32_f16(false, (a), false, (b), (short)0, (c), false, false)

static __device__ __forceinline__ v16h pack2(v8h lo, v8h hi) {
  v16h r;
#pragma unroll
  for (int i = 0; i < 8; ++i) { r[i] = lo[i]; r[8 + i] = hi[i]; }
  return r;
}

// A fragment (16x32, f16) from row-major f16 source. base -> (row0, k0).
// Lane L: row = L%16; halves j -> k = 16*(L/16)+j  (16 contiguous halves per lane)
static __device__ __forceinline__ v16h load_a(const _Float16* base, int stride, int lane) {
  const int r  = lane & 15;
  const int kb = (lane >> 4) * 16;
  const _Float16* p = base + (size_t)r * stride + kb;
  return pack2(*(const v8h*)p, *(const v8h*)(p + 8));
}

// B fragment (32x16): B(kd, n) = src[n][kd] where src is row-major with `stride`.
// Lane L: n = L%16 ; halves j -> kd = 16*(L/16) + j  (16 contiguous halves per lane)
static __device__ __forceinline__ v16h load_bt(const _Float16* base, int stride, int lane) {
  const int n  = lane & 15;
  const int kb = (lane >> 4) * 16;
  const _Float16* p = base + (size_t)n * stride + kb;
  return pack2(*(const v8h*)p, *(const v8h*)(p + 8));
}

static __device__ __forceinline__ float halfwave_max(float v) {
#pragma unroll
  for (int off = 8; off >= 1; off >>= 1) v = fmaxf(v, __shfl_xor(v, off, 32));
  return v;
}
static __device__ __forceinline__ float halfwave_sum(float v) {
#pragma unroll
  for (int off = 8; off >= 1; off >>= 1) v += __shfl_xor(v, off, 32);
  return v;
}

// ---------------------------------------------------------------------------
// f32 -> f16 conversion
// ---------------------------------------------------------------------------
__global__ void __launch_bounds__(256) cvt_f16_kernel(const float* __restrict__ in,
                                                      _Float16* __restrict__ out, int n) {
  int i = blockIdx.x * 256 + threadIdx.x;
  if (i < n) out[i] = (_Float16)in[i];
}

// ---------------------------------------------------------------------------
// Pack W [K,N] (f32, row-major) into WMMA B-fragments (f16).
// Fragment f = ntile*(K/32)+ktile holds 32 lanes x 16 halves, contiguous 1KB.
// Element (lane, j): k = ktile*32 + 16*(lane/16) + j, n = ntile*16 + lane%16.
// ---------------------------------------------------------------------------
__global__ void __launch_bounds__(256) pack_w_kernel(const float* __restrict__ W,
                                                     _Float16* __restrict__ P,
                                                     int K, int N) {
  size_t t = (size_t)blockIdx.x * 256 + threadIdx.x;
  if (t >= (size_t)K * N) return;
  int j    = (int)(t & 15);
  int lane = (int)((t >> 4) & 31);
  size_t f = t >> 9;
  int ktiles = K >> 5;
  int ntile = (int)(f / ktiles);
  int ktile = (int)(f % ktiles);
  int k = ktile * 32 + ((lane >> 4) << 4) + j;
  int n = ntile * 16 + (lane & 15);
  P[t] = (_Float16)W[(size_t)k * N + n];
}

// ---------------------------------------------------------------------------
// Transpose V section of qkv [8192,3072] into VT[b*H+h][hd][s] (f16).
// ---------------------------------------------------------------------------
__global__ void __launch_bounds__(256) transpose_v_kernel(const _Float16* __restrict__ qkv,
                                                          _Float16* __restrict__ vt) {
  size_t t = (size_t)blockIdx.x * 256 + threadIdx.x;   // total 4*16*64*2048
  int s = (int)(t & 2047);
  size_t rest = t >> 11;
  int d = (int)(rest & 63);
  int bh = (int)(rest >> 6);          // 0..63
  int h = bh & 15, b = bh >> 4;
  vt[t] = qkv[((size_t)(b * 2048 + s)) * 3072 + 2048 + h * 64 + d];
}

// ---------------------------------------------------------------------------
// GEMM (f16 in, f16 out):  C[M,N] = A[M,K] * W[K,N]
// Block = 256 thr = 8 waves arranged 2(M) x 4(N).
// Wave computes 32x64 (2 A-frags x 4 B-frags = 8 WMMAs / 32-k step);
// each B fragment is reused by both A fragments from registers.
// ---------------------------------------------------------------------------
__global__ void __launch_bounds__(256) gemm_f16_kernel(const _Float16* __restrict__ A,
                                                       const _Float16* __restrict__ Bp,
                                                       _Float16* __restrict__ C,
                                                       int K, int N) {
  const int lane = threadIdx.x & 31;
  const int wave = threadIdx.x >> 5;
  const int m0 = blockIdx.x * 64 + (wave & 1) * 32;
  const int n0 = blockIdx.y * 256 + (wave >> 1) * 64;
  const int ktiles = K >> 5;

  v8f aclo[4] = {{}, {}, {}, {}};   // rows m0..m0+15
  v8f achi[4] = {{}, {}, {}, {}};   // rows m0+16..m0+31
  const _Float16* abase0 = A + (size_t)m0 * K;
  const _Float16* abase1 = abase0 + (size_t)16 * K;
  const size_t fstride = (size_t)ktiles * 32 * 16;   // halves per n-tile column

  for (int k = 0; k < K; k += 32) {
    v16h a0 = load_a(abase0 + k, K, lane);
    v16h a1 = load_a(abase1 + k, K, lane);
    const _Float16* bp = Bp + (((size_t)(n0 >> 4) * ktiles + (k >> 5)) * 32 + lane) * 16;
#pragma unroll
    for (int t = 0; t < 4; ++t) {
      v16h b = *(const v16h*)(bp + (size_t)t * fstride);
      aclo[t] = WMMA_F32_F16(a0, b, aclo[t]);
      achi[t] = WMMA_F32_F16(a1, b, achi[t]);
    }
  }

  const int rlo = m0 + ((lane >> 4) << 3);
  const int c0 = n0 + (lane & 15);
#pragma unroll
  for (int j = 0; j < 8; ++j) {
    _Float16* row = C + (size_t)(rlo + j) * N + c0;
    row[0]  = (_Float16)aclo[0][j];
    row[16] = (_Float16)aclo[1][j];
    row[32] = (_Float16)aclo[2][j];
    row[48] = (_Float16)aclo[3][j];
    _Float16* row2 = row + (size_t)16 * N;
    row2[0]  = (_Float16)achi[0][j];
    row2[16] = (_Float16)achi[1][j];
    row2[32] = (_Float16)achi[2][j];
    row2[48] = (_Float16)achi[3][j];
  }
}

// ---------------------------------------------------------------------------
// GEMM (f16 in, f32 out + bias):  C[M,N] = A[M,K] * W[K,N] + bias[N]
// Same 32x64 per-wave blocking.
// ---------------------------------------------------------------------------
__global__ void __launch_bounds__(256) gemm_bias_f32_kernel(const _Float16* __restrict__ A,
                                                            const _Float16* __restrict__ Bp,
                                                            const float* __restrict__ bias,
                                                            float* __restrict__ C,
                                                            int K, int N) {
  const int lane = threadIdx.x & 31;
  const int wave = threadIdx.x >> 5;
  const int m0 = blockIdx.x * 64 + (wave & 1) * 32;
  const int n0 = blockIdx.y * 256 + (wave >> 1) * 64;
  const int ktiles = K >> 5;

  v8f aclo[4] = {{}, {}, {}, {}};
  v8f achi[4] = {{}, {}, {}, {}};
  const _Float16* abase0 = A + (size_t)m0 * K;
  const _Float16* abase1 = abase0 + (size_t)16 * K;
  const size_t fstride = (size_t)ktiles * 32 * 16;

  for (int k = 0; k < K; k += 32) {
    v16h a0 = load_a(abase0 + k, K, lane);
    v16h a1 = load_a(abase1 + k, K, lane);
    const _Float16* bp = Bp + (((size_t)(n0 >> 4) * ktiles + (k >> 5)) * 32 + lane) * 16;
#pragma unroll
    for (int t = 0; t < 4; ++t) {
      v16h b = *(const v16h*)(bp + (size_t)t * fstride);
      aclo[t] = WMMA_F32_F16(a0, b, aclo[t]);
      achi[t] = WMMA_F32_F16(a1, b, achi[t]);
    }
  }

  const int rlo = m0 + ((lane >> 4) << 3);
  const int c0 = n0 + (lane & 15);
  const float bb0 = bias[c0];
  const float bb1 = bias[c0 + 16];
  const float bb2 = bias[c0 + 32];
  const float bb3 = bias[c0 + 48];
#pragma unroll
  for (int j = 0; j < 8; ++j) {
    float* row = C + (size_t)(rlo + j) * N + c0;
    row[0]  = aclo[0][j] + bb0;
    row[16] = aclo[1][j] + bb1;
    row[32] = aclo[2][j] + bb2;
    row[48] = aclo[3][j] + bb3;
    float* row2 = row + (size_t)16 * N;
    row2[0]  = achi[0][j] + bb0;
    row2[16] = achi[1][j] + bb1;
    row2[32] = achi[2][j] + bb2;
    row2[48] = achi[3][j] + bb3;
  }
}

// ---------------------------------------------------------------------------
// Flash attention (causal). One wave handles 16 query rows of one (b,h).
// Per 32-key step: 4 WMMAs for S = Q K^T (f32), online softmax, P transposed
// C-layout -> A-layout via per-wave LDS tile, 4 WMMAs for O += P V.
// qkv: [8192, 3072] f16 (q|k|v), vt: [b*H+h][hd][s] f16, O: [8192,1024] f16
// ---------------------------------------------------------------------------
__global__ void __launch_bounds__(256) attn_fwd_kernel(const _Float16* __restrict__ qkv,
                                                       const _Float16* __restrict__ vt,
                                                       _Float16* __restrict__ O) {
  const int S = 2048, NQ = 3072;
  const int lane = threadIdx.x & 31;
  const int wave = threadIdx.x >> 5;
  const int blk = blockIdx.x;           // 1024 blocks = 64 (b,h) * 16
  const int bh = blk >> 4;
  const int b = bh >> 4, h = bh & 15;
  const int q0 = ((blk & 15) * 8 + wave) * 16;

  const _Float16* Qb = qkv + (size_t)b * S * NQ + (size_t)h * 64;
  const _Float16* Kb = Qb + 1024;
  const _Float16* Vt = vt + (size_t)bh * 64 * S;

  __shared__ __align__(16) _Float16 plds[8][16][40];  // per-wave 16x32 P tile (padded)

  // Q fragments for hd 0..31 and 32..63, reused across all key tiles
  const v16h aq0 = load_a(Qb + (size_t)q0 * NQ, NQ, lane);
  const v16h aq1 = load_a(Qb + (size_t)q0 * NQ + 32, NQ, lane);

  float m[8], l[8];
#pragma unroll
  for (int j = 0; j < 8; ++j) { m[j] = -1e30f; l[j] = 0.0f; }
  v8f o0 = {}, o1 = {}, o2 = {}, o3 = {};

  const int half  = lane >> 4;       // 0 or 1
  const int nn    = lane & 15;
  const int rbase = q0 + 8 * half;   // query row of C-layout vgpr j = rbase + j

  for (int kt = 0; kt < q0 + 16; kt += 32) {
    // ---- scores: S[16 q x 32 k] = Q (16x64) * K^T (64x32), f32 accum ----
    v8f s0 = {}, s1 = {};
    {
      const _Float16* k0 = Kb + (size_t)kt * NQ;
      v16h b00 = load_bt(k0, NQ, lane);        // keys kt..kt+15, hd 0..31
      v16h b01 = load_bt(k0 + 32, NQ, lane);   // keys kt..kt+15, hd 32..63
      s0 = WMMA_F32_F16(aq0, b00, s0);
      s0 = WMMA_F32_F16(aq1, b01, s0);
      const _Float16* k1 = Kb + (size_t)(kt + 16) * NQ;
      v16h b10 = load_bt(k1, NQ, lane);        // keys kt+16..kt+31
      v16h b11 = load_bt(k1 + 32, NQ, lane);
      s1 = WMMA_F32_F16(aq0, b10, s1);
      s1 = WMMA_F32_F16(aq1, b11, s1);
    }

    // ---- scale, causal mask, online softmax (row = across 16 lanes/half) ----
    float alpha[8];
#pragma unroll
    for (int j = 0; j < 8; ++j) {
      const int row = rbase + j;
      float x0 = s0[j] * 0.125f;               // HD^-0.5
      float x1 = s1[j] * 0.125f;
      if (kt + nn > row)      x0 = -1e30f;     // key index kt+nn
      if (kt + 16 + nn > row) x1 = -1e30f;
      const float t  = halfwave_max(fmaxf(x0, x1));
      const float mn = fmaxf(m[j], t);
      alpha[j] = __expf(m[j] - mn);
      m[j] = mn;
      const float p0 = __expf(x0 - mn);
      const float p1 = __expf(x1 - mn);
      l[j] = l[j] * alpha[j] + halfwave_sum(p0 + p1);
      plds[wave][8 * half + j][nn]      = (_Float16)p0;   // C-layout -> LDS
      plds[wave][8 * half + j][16 + nn] = (_Float16)p1;
    }
    __builtin_amdgcn_wave_barrier();   // wave-internal DS ordering is in-order

    // ---- read P back in A-fragment layout ----
    v16h ap;
    {
      const _Float16* pr = &plds[wave][nn][0];     // row = lane%16
      const int kb8 = half * 16;
      ap = pack2(*(const v8h*)(pr + kb8), *(const v8h*)(pr + 8 + kb8));
    }

    // ---- rescale O and accumulate O += P * V (K-dim = 32 keys) ----
#pragma unroll
    for (int j = 0; j < 8; ++j) {
      o0[j] *= alpha[j]; o1[j] *= alpha[j]; o2[j] *= alpha[j]; o3[j] *= alpha[j];
    }
    v16h bv0 = load_bt(Vt + kt, S, lane);                    // hd cols  0..15
    v16h bv1 = load_bt(Vt + (size_t)16 * S + kt, S, lane);   // hd cols 16..31
    v16h bv2 = load_bt(Vt + (size_t)32 * S + kt, S, lane);   // hd cols 32..47
    v16h bv3 = load_bt(Vt + (size_t)48 * S + kt, S, lane);   // hd cols 48..63
    o0 = WMMA_F32_F16(ap, bv0, o0);
    o1 = WMMA_F32_F16(ap, bv1, o1);
    o2 = WMMA_F32_F16(ap, bv2, o2);
    o3 = WMMA_F32_F16(ap, bv3, o3);
  }

  // ---- normalize by row sums, store to O [8192, 1024] (head-interleaved) ----
  _Float16* Ob = O + (size_t)(b * S + rbase) * 1024 + h * 64 + nn;
#pragma unroll
  for (int j = 0; j < 8; ++j) {
    const float inv = 1.0f / l[j];
    _Float16* row = Ob + (size_t)j * 1024;
    row[0]  = (_Float16)(o0[j] * inv);
    row[16] = (_Float16)(o1[j] * inv);
    row[32] = (_Float16)(o2[j] * inv);
    row[48] = (_Float16)(o3[j] * inv);
  }
}

// ---------------------------------------------------------------------------
// Host launcher
// ---------------------------------------------------------------------------
extern "C" void kernel_launch(void* const* d_in, const int* in_sizes, int n_in,
                              void* d_out, int out_size, void* d_ws, size_t ws_size,
                              hipStream_t stream) {
  (void)in_sizes; (void)n_in; (void)out_size; (void)ws_size;
  const float* x     = (const float*)d_in[0];   // [4,2048,1024]
  const float* w_qkv = (const float*)d_in[1];   // [1024,3072]
  const float* w_out = (const float*)d_in[2];   // [1024,1024]
  const float* b_out = (const float*)d_in[3];   // [1024]
  float* out = (float*)d_out;                   // [4,2048,1024]

  const int M = 8192;     // B*S
  char* ws = (char*)d_ws;
  size_t off = 0;
  auto take = [&](size_t bytes) -> char* {
    char* p = ws + off;
    off += (bytes + 255) & ~(size_t)255;
    return p;
  };
  _Float16* xh   = (_Float16*)take((size_t)M * 1024 * 2);      // x in f16
  _Float16* wqp  = (_Float16*)take((size_t)1024 * 3072 * 2);   // packed qkv weights
  _Float16* wop  = (_Float16*)take((size_t)1024 * 1024 * 2);   // packed out weights
  _Float16* qkvh = (_Float16*)take((size_t)M * 3072 * 2);      // qkv activations f16
  _Float16* vth  = (_Float16*)take((size_t)4096 * 2048 * 2);   // V^T per (b,h)
  _Float16* oh   = (_Float16*)take((size_t)M * 1024 * 2);      // attention output f16

  // 1) precision conversion + weight packing into WMMA B-fragment layout
  cvt_f16_kernel<<<(M * 1024) / 256, 256, 0, stream>>>(x, xh, M * 1024);
  pack_w_kernel<<<(1024 * 3072) / 256, 256, 0, stream>>>(w_qkv, wqp, 1024, 3072);
  pack_w_kernel<<<(1024 * 1024) / 256, 256, 0, stream>>>(w_out, wop, 1024, 1024);

  // 2) qkv = x @ w_qkv   (M=8192, N=3072, K=1024)
  gemm_f16_kernel<<<dim3(M / 64, 3072 / 256), 256, 0, stream>>>(xh, wqp, qkvh, 1024, 3072);

  // 3) V^T per (b,h) for contiguous P*V B-fragments
  transpose_v_kernel<<<(4096 * 2048) / 256, 256, 0, stream>>>(qkvh, vth);

  // 4) causal flash attention, WMMA everywhere
  attn_fwd_kernel<<<1024, 256, 0, stream>>>(qkvh, vth, oh);

  // 5) out = attn @ w_out + b_out   (M=8192, N=1024, K=1024), f32 output
  gemm_bias_f32_kernel<<<dim3(M / 64, 1024 / 256), 256, 0, stream>>>(oh, wop, b_out, out, 1024, 1024);
}
